// AngularDistribution_2774548873938
// MI455X (gfx1250) — matrix operand order repaired and checked by reading, not compiled, not run on hardware
//
#include <hip/hip_runtime.h>

// ---------------------------------------------------------------------------
// Angular (Behler) symmetry-function distribution for MI455X (gfx1250).
//
// Per (batch, atom):  OUT[32 x 8] = RAD^T[32 x 2048] * ANG[2048 x 8]
// computed as f16 WMMA (v_wmma_f32_16x16x32_f16) with f32 accumulation.
// 512 blocks (one per (b,a)), 256 threads = 8 wave32 waves, each wave owns
// 8 K-chunks of 32 triples; cross-wave reduction in LDS (deterministic).
// ---------------------------------------------------------------------------

typedef __attribute__((ext_vector_type(16))) _Float16 v16h;
typedef __attribute__((ext_vector_type(8)))  float    v8f;
typedef __attribute__((ext_vector_type(4)))  float    v4f;

#define K_GAMMA  4.0f
#define K_CUTOFF 5.0f

constexpr int B_ = 4, A_ = 128, N_ = 2048;
constexpr int WAVES = 8;                 // 256 threads / wave32
constexpr int CHUNK = 32;                // K per WMMA
constexpr int CHUNKS_PER_BA  = N_ / CHUNK;          // 64
constexpr int CHUNKS_PER_WAVE = CHUNKS_PER_BA / WAVES; // 8

__global__ __launch_bounds__(256)
void angdist_wmma_kernel(const float* __restrict__ r_ij,
                         const float* __restrict__ r_ik,
                         const float* __restrict__ r_jk,
                         const float* __restrict__ offsets,
                         const int*   __restrict__ masks,
                         float*       __restrict__ out)
{
    const int ba   = blockIdx.x;            // 0..511  (b*128 + a)
    const int tid  = threadIdx.x;
    const int wave = tid >> 5;
    const int lane = tid & 31;
    const int m    = lane & 15;             // A row within tile / B column
    const long base = (long)ba * N_;

    // wave-private triple scalars: {a', b', cos_theta, cut*mask} per triple
    __shared__ v4f   s_trip[WAVES][CHUNK];             // 4 KB
    __shared__ float s_red[WAVES * 2 * 8 * 32];        // 16 KB wave partials

    const float L2E = 1.44269504088896340736f;         // log2(e)
    // Per-lane Gaussian centers for the two r-tiles (rows m and m+16).
    const float o0 = offsets[m];
    const float o1 = offsets[m + 16];
    const float t0 = -3.0f * K_GAMMA * L2E * o0 * o0;  // -3*gamma*o^2 (log2 dom.)
    const float t1 = -3.0f * K_GAMMA * L2E * o1 * o1;

    // B-fragment lane config: column f = m (zero for f >= 8).
    const bool  bzero  = (m >= 8);
    const int   f      = m & 7;
    const int   zlog   = f & 3;             // zeta = 1 << zlog  (1,2,4,8)
    const bool  negf   = (f >= 4);          // (1 + cos) branch
    const float bscale = (zlog == 0) ? 1.0f
                       : (zlog == 1) ? 0.5f
                       : (zlog == 2) ? 0.125f
                       :               0.0078125f;     // 2^(1-zeta)

    v8f acc0 = {};  // rows r = 0..15
    v8f acc1 = {};  // rows r = 16..31

    for (int c = 0; c < CHUNKS_PER_WAVE; ++c) {
        const int  chunk = wave + c * WAVES;
        const long n     = base + (long)chunk * CHUNK + lane;

        const float rij = r_ij[n];
        const float rik = r_ik[n];
        const float rjk = r_jk[n];
        const int   mk  = masks[n];

        if (c + 1 < CHUNKS_PER_WAVE) {
            const long np = n + (long)WAVES * CHUNK;
            __builtin_prefetch(&r_ij[np], 0, 1);       // global_prefetch_b8
            __builtin_prefetch(&r_ik[np], 0, 1);
            __builtin_prefetch(&r_jk[np], 0, 1);
            __builtin_prefetch(&masks[np], 0, 1);
        }

        // Folded Gaussian exponent:  -g*((rij-o)^2+(rik-o)^2+(rjk-o)^2)
        //   = (-g*s2) + (2g*s1)*o + (-3g)*o^2      (converted to log2 domain)
        const float s1 = rij + rik + rjk;
        const float q_ij = rij * rij, q_ik = rik * rik, q_jk = rjk * rjk;
        const float s2 = q_ij + q_ik + q_jk;

        float ct = (q_ij + q_ik - q_jk) / (2.0f * rij * rik);
        ct = (mk != 0) ? ct : 0.0f;

        // cosine cutoff product (r in [0.5,5) so branch is nearly uniform)
        const float kpi = 3.14159265358979323846f / K_CUTOFF;
        float cij = 0.5f * (__cosf(rij * kpi) + 1.0f); cij = (rij < K_CUTOFF) ? cij : 0.0f;
        float cik = 0.5f * (__cosf(rik * kpi) + 1.0f); cik = (rik < K_CUTOFF) ? cik : 0.0f;
        float cjk = 0.5f * (__cosf(rjk * kpi) + 1.0f); cjk = (rjk < K_CUTOFF) ? cjk : 0.0f;
        const float cm = (mk != 0) ? (cij * cik * cjk) : 0.0f;

        v4f tp;
        tp.x = -K_GAMMA * L2E * s2;          // a'
        tp.y =  2.0f * K_GAMMA * L2E * s1;   // b'
        tp.z = ct;
        tp.w = cm;
        s_trip[wave][lane] = tp;
        // Producer/consumer within one wave: LDS ops are issued and complete
        // in order for a wave (DScnt in-order), no barrier required.

        // ---- build WMMA fragments -------------------------------------
        v16h a0, a1, bf;
        #pragma unroll
        for (int i = 0; i < 16; ++i) {
            // A-fragment k index (ISA 16-bit A 16x32 layout):
            //   lane<16: K = {0..7, 16..23};  lane>=16: K = {8..15, 24..31}
            const int kA = (lane < 16) ? ((i < 8) ? i : i + 8)
                                       : ((i < 8) ? i + 8 : i + 16);
            const v4f p = s_trip[wave][kA];
            const float e0 = __builtin_fmaf(p.y, o0, p.x) + t0;
            const float e1 = __builtin_fmaf(p.y, o1, p.x) + t1;
            a0[i] = (_Float16)(p.w * exp2f(e0));       // v_exp_f32
            a1[i] = (_Float16)(p.w * exp2f(e1));

            // B-fragment k index: lane<16: K = {0..7, 16..23}; lane>=16: K = {8..15, 24..31}
            const int kB = kA;
            const v4f q = s_trip[wave][kB];
            const float x  = negf ? (1.0f + q.z) : (1.0f - q.z);
            const float x2 = x * x, x4 = x2 * x2, x8 = x4 * x4;
            const float xv = (zlog == 0) ? x
                           : (zlog == 1) ? x2
                           : (zlog == 2) ? x4
                           :               x8;
            bf[i] = bzero ? (_Float16)0.0f : (_Float16)(xv * bscale);
        }

        // D = A x B + C   (16x16x32 f16 -> f32)
        acc0 = __builtin_amdgcn_wmma_f32_16x16x32_f16(
                   false, a0, false, bf, (short)0, acc0, false, false);
        acc1 = __builtin_amdgcn_wmma_f32_16x16x32_f16(
                   false, a1, false, bf, (short)0, acc1, false, false);
    }

    // ---- deterministic cross-wave reduction in LDS ---------------------
    #pragma unroll
    for (int j = 0; j < 8; ++j) {
        s_red[((wave * 2 + 0) * 8 + j) * 32 + lane] = acc0[j];
        s_red[((wave * 2 + 1) * 8 + j) * 32 + lane] = acc1[j];
    }
    __syncthreads();

    // 256 threads -> 256 (r,f) outputs. C/D layout: VGPR j, lane l holds
    // element (M = j + 8*(l>=16), N = l%16); only N < 8 is meaningful.
    {
        const int tile = tid >> 7;           // 0..1
        const int rem  = tid & 127;
        const int j    = rem >> 4;           // 0..7 (VGPR index)
        const int idx  = rem & 15;
        const int ln   = (idx < 8) ? idx : idx + 8;   // lanes {0..7,16..23}
        const int fo   = idx & 7;
        const int r    = tile * 16 + j + ((idx >= 8) ? 8 : 0);

        float sum = 0.0f;
        #pragma unroll
        for (int w = 0; w < WAVES; ++w)
            sum += s_red[((w * 2 + tile) * 8 + j) * 32 + ln];

        out[(long)ba * 256 + r * 8 + fo] = sum;
    }
}

extern "C" void kernel_launch(void* const* d_in, const int* in_sizes, int n_in,
                              void* d_out, int out_size, void* d_ws, size_t ws_size,
                              hipStream_t stream)
{
    (void)in_sizes; (void)n_in; (void)out_size; (void)d_ws; (void)ws_size;
    const float* r_ij    = (const float*)d_in[0];
    const float* r_ik    = (const float*)d_in[1];
    const float* r_jk    = (const float*)d_in[2];
    const float* offsets = (const float*)d_in[3];
    const int*   masks   = (const int*)d_in[4];
    float*       out     = (float*)d_out;

    dim3 grid(B_ * A_);     // 512 (b,a) tiles
    dim3 block(256);        // 8 wave32 waves
    angdist_wmma_kernel<<<grid, block, 0, stream>>>(r_ij, r_ik, r_jk,
                                                    offsets, masks, out);
}
